// Net_70643622085082
// MI455X (gfx1250) — compile-verified, matrix operand order checked by reading, not run on hardware
//
#include <hip/hip_runtime.h>
#include <hip/hip_bf16.h>

#define EMB     300
#define TOWERS  6
#define FO      50            // F_OUT = EMB / TOWERS
#define NLAYERS 4
#define EPSV    1e-5f
#define KPE     320           // 300 padded to mult of 32
#define KP3     928           // 900 padded
#define AGGW    1200          // [mean, min, max, std] * 300
#define AGGP    1216          // 1200 padded
#define KP2E    608           // 600 padded

typedef __bf16 bf16;
typedef __attribute__((ext_vector_type(16))) __bf16 v16bf;
typedef __attribute__((ext_vector_type(8)))  __bf16 v8bf;
typedef __attribute__((ext_vector_type(8)))  float  v8f;

// ---------------------------------------------------------------- helpers
__device__ __forceinline__ float bn1(float x, float g, float b, float m, float v) {
    return (x - m) * rsqrtf(v + EPSV) * g + b;
}

__device__ __forceinline__ void atomicMaxF(float* addr, float val) {
    unsigned int* ua = (unsigned int*)addr;
    unsigned int old = *ua;
    while (__uint_as_float(old) < val) {
        unsigned int assumed = old;
        old = atomicCAS(ua, assumed, __float_as_uint(val));
        if (old == assumed) break;
    }
}
__device__ __forceinline__ void atomicMinF(float* addr, float val) {
    unsigned int* ua = (unsigned int*)addr;
    unsigned int old = *ua;
    while (__uint_as_float(old) > val) {
        unsigned int assumed = old;
        old = atomicCAS(ua, assumed, __float_as_uint(val));
        if (old == assumed) break;
    }
}

// Fragment load: per-lane halves 0..7 <- K = base..base+7, halves 8..15 <- K = base+16..+23
__device__ __forceinline__ v16bf ldfrag(const bf16* p) {
    v8bf lo = *(const v8bf*)(p);
    v8bf hi = *(const v8bf*)(p + 16);
    return __builtin_shufflevector(lo, hi, 0, 1, 2, 3, 4, 5, 6, 7,
                                   8, 9, 10, 11, 12, 13, 14, 15);
}

// ---------------------------------------------------------------- elementwise
__global__ void k_fill(float* p, long n, float val) {
    long i = blockIdx.x * (long)blockDim.x + threadIdx.x;
    long s = (long)gridDim.x * blockDim.x;
    for (; i < n; i += s) p[i] = val;
}

__global__ void k_add(float* __restrict__ a, const float* __restrict__ b, long n) {
    long i = blockIdx.x * (long)blockDim.x + threadIdx.x;
    long s = (long)gridDim.x * blockDim.x;
    for (; i < n; i += s) a[i] += b[i];
}

// f32 (rows x K) -> bf16 (rows x KP), zero pad K..KP
__global__ void k_f2bf_pad(const float* __restrict__ in, bf16* __restrict__ out,
                           int K, int KP, long n /* rows*KP */) {
    long i = blockIdx.x * (long)blockDim.x + threadIdx.x;
    long s = (long)gridDim.x * blockDim.x;
    for (; i < n; i += s) {
        long r = i / KP; int k = (int)(i % KP);
        out[i] = (k < K) ? (bf16)in[r * K + k] : (bf16)0.0f;
    }
}

// Weight transpose+pad: W (batch x Ksrc x N, f32, rows rowOff..rowOff+K used)
// -> out (batch x N x KP, bf16), zero pad K..KP
__global__ void k_wconvT(const float* __restrict__ W, bf16* __restrict__ out,
                         int Ksrc, int rowOff, int K, int N, int KP,
                         long n /* batch*N*KP */) {
    long i = blockIdx.x * (long)blockDim.x + threadIdx.x;
    long s = (long)gridDim.x * blockDim.x;
    long perB = (long)N * KP;
    for (; i < n; i += s) {
        long b = i / perB;
        long nk = i % perB;
        int nn = (int)(nk / KP);
        int k  = (int)(nk % KP);
        float v = (k < K) ? W[b * (long)Ksrc * N + (long)(rowOff + k) * N + nn] : 0.0f;
        out[i] = (bf16)v;
    }
}

__global__ void k_embed(const int* __restrict__ idx, const float* __restrict__ tab,
                        float* __restrict__ out, long n /* rows*EMB */) {
    long i = blockIdx.x * (long)blockDim.x + threadIdx.x;
    long s = (long)gridDim.x * blockDim.x;
    for (; i < n; i += s) {
        long r = i / EMB; int d = (int)(i % EMB);
        out[i] = tab[(long)idx[r] * EMB + d];
    }
}

// ---------------------------------------------------------------- graph stats
__global__ void k_count(const int* __restrict__ dst, float* cnt, int E) {
    long i = blockIdx.x * (long)blockDim.x + threadIdx.x;
    long s = (long)gridDim.x * blockDim.x;
    for (; i < E; i += s) atomicAdd(&cnt[dst[i]], 1.0f);
}

__global__ void k_logsum(const float* __restrict__ cnt, float* scal, int N) {
    long i = blockIdx.x * (long)blockDim.x + threadIdx.x;
    long s = (long)gridDim.x * blockDim.x;
    for (; i < N; i += s) atomicAdd(scal, logf(cnt[i] + 1.0f));
}

__global__ void k_node_stats(const float* __restrict__ cnt, const float* __restrict__ scal,
                             float* deg, float* amp, float* att, float* hnb, int N) {
    long i = blockIdx.x * (long)blockDim.x + threadIdx.x;
    long s = (long)gridDim.x * blockDim.x;
    float avg = scal[0] / (float)N;
    for (; i < N; i += s) {
        float c = cnt[i];
        float d = fmaxf(c, 1.0f);
        float ld = logf(d + 1.0f);
        deg[i] = d;
        amp[i] = ld / avg;
        att[i] = avg / ld;
        hnb[i] = (c > 0.0f) ? 1.0f : 0.0f;
    }
}

// ---------------------------------------------------------------- message pipeline
// hcat: E x KP3, [h[dst] | h[src] | eproj | 0-pad]
__global__ void k_build_hcat(const float* __restrict__ h, const int* __restrict__ src,
                             const int* __restrict__ dst, const float* __restrict__ eproj,
                             bf16* __restrict__ hcat, long n /* E*KP3 */) {
    long i = blockIdx.x * (long)blockDim.x + threadIdx.x;
    long s = (long)gridDim.x * blockDim.x;
    for (; i < n; i += s) {
        long e = i / KP3; int d = (int)(i % KP3);
        float v;
        if (d < EMB)              v = h[(long)dst[e] * EMB + d];
        else if (d < 2 * EMB)     v = h[(long)src[e] * EMB + (d - EMB)];
        else if (d < 3 * EMB)     v = eproj[e * (long)EMB + (d - 2 * EMB)];
        else                      v = 0.0f;
        hcat[i] = (bf16)v;
    }
}

__global__ void k_scatter_stats(const float* __restrict__ m, const int* __restrict__ dst,
                                float* s, float* mn, float* mx, float* sq,
                                int E, int N, long n /* T*E*EMB */) {
    long i = blockIdx.x * (long)blockDim.x + threadIdx.x;
    long st = (long)gridDim.x * blockDim.x;
    for (; i < n; i += st) {
        int d = (int)(i % EMB);
        long te = i / EMB;
        int e = (int)(te % E);
        int t = (int)(te / E);
        float v = m[i];
        long base = ((long)t * N + dst[e]) * EMB + d;
        atomicAdd(&s[base], v);
        atomicAdd(&sq[base], v * v);
        atomicMaxF(&mx[base], v);
        atomicMinF(&mn[base], v);
    }
}

// agg: (T x N x AGGP) bf16, full padded rows written
__global__ void k_finalize_agg(const float* __restrict__ s, const float* __restrict__ mn,
                               const float* __restrict__ mx, const float* __restrict__ sq,
                               const float* __restrict__ deg, const float* __restrict__ hnb,
                               bf16* __restrict__ agg, int N, long n /* T*N*AGGP */) {
    long i = blockIdx.x * (long)blockDim.x + threadIdx.x;
    long st = (long)gridDim.x * blockDim.x;
    for (; i < n; i += st) {
        int d = (int)(i % AGGP);
        long tn = i / AGGP;
        int nd = (int)(tn % N);
        long t  = tn / N;
        float v = 0.0f;
        if (d < AGGW) {
            int seg = d / EMB, dd = d % EMB;
            long sidx = ((long)t * N + nd) * EMB + dd;
            float dg = deg[nd];
            float hb = hnb[nd];
            if (seg == 0) {
                v = s[sidx] / dg;
            } else if (seg == 1) {
                v = (hb > 0.0f) ? mn[sidx] : 0.0f;
            } else if (seg == 2) {
                v = (hb > 0.0f) ? mx[sidx] : 0.0f;
            } else {
                float mean = s[sidx] / dg;
                float var  = sq[sidx] / dg - mean * mean;
                v = sqrtf(fmaxf(var, 0.0f) + EPSV);
            }
        }
        agg[i] = (bf16)v;
    }
}

// out[n, t*FO+o] = hA + P1 + amp[n]*P2 + att[n]*P3 + b_post[t,o]
__global__ void k_combine_post(const float* __restrict__ tmp, const float* __restrict__ amp,
                               const float* __restrict__ att, const float* __restrict__ bpost,
                               float* __restrict__ out, int N, long n /* N*T*FO */) {
    long i = blockIdx.x * (long)blockDim.x + threadIdx.x;
    long st = (long)gridDim.x * blockDim.x;
    long TNF = (long)TOWERS * N * FO;
    for (; i < n; i += st) {
        int nd  = (int)(i / (TOWERS * FO));
        int rem = (int)(i % (TOWERS * FO));
        int t = rem / FO, o = rem % FO;
        long idx = ((long)t * N + nd) * FO + o;
        float v = tmp[idx] + tmp[TNF + idx]
                + amp[nd] * tmp[2 * TNF + idx]
                + att[nd] * tmp[3 * TNF + idx]
                + bpost[t * FO + o];
        out[(long)nd * EMB + t * FO + o] = v;
    }
}

__global__ void k_segsum(const float* __restrict__ nrep, const int* __restrict__ batch,
                         float* __restrict__ g, long n /* N*EMB */) {
    long i = blockIdx.x * (long)blockDim.x + threadIdx.x;
    long st = (long)gridDim.x * blockDim.x;
    for (; i < n; i += st) {
        long nd = i / EMB; int d = (int)(i % EMB);
        atomicAdd(&g[(long)batch[nd] * EMB + d], nrep[i]);
    }
}

// ---------------------------------------------------------------- WMMA GEMM
// C[bz](MxN,f32) = A[bz](MxKP bf16, zero-padded) * BT[bz](NxKP bf16, transposed+padded)
// 2x2 16x16 tiles per wave (32x32 of C), guard-free inner loop (clamped edge rows).
// epi: bit0 = relu, bit1 = batchnorm
#define GEMM_WAVES 4
__global__ void k_gemm_bf16(const bf16* __restrict__ A, long sA,
                            const bf16* __restrict__ BT, long sB,
                            const float* __restrict__ bias, long sBias,
                            float* __restrict__ C, long sC, int ldc,
                            int M, int N, int KP, int epi,
                            const float* __restrict__ bng, const float* __restrict__ bnb,
                            const float* __restrict__ bnm, const float* __restrict__ bnv) {
    const int lane = threadIdx.x & 31;
    const int wv   = threadIdx.x >> 5;
    const int nt = blockIdx.x;
    const int mt = blockIdx.y * GEMM_WAVES + wv;
    if (mt * 32 >= M) return;               // wave-uniform exit; EXEC all-1 for WMMA
    const int bz = blockIdx.z;
    A  += (long)bz * sA;
    BT += (long)bz * sB;
    C  += (long)bz * sC;
    const float* biasb = bias ? bias + (long)bz * sBias : nullptr;

    const int g  = lane >> 4;               // half-wave K-striping group (ISA 7.12.2)
    const int ml = lane & 15;
    const int Mc = M - 1, Nc = N - 1;
    int ar0 = mt * 32 + ml;       ar0 = ar0 < Mc ? ar0 : Mc;
    int ar1 = mt * 32 + 16 + ml;  ar1 = ar1 < Mc ? ar1 : Mc;
    int bc0 = nt * 32 + ml;       bc0 = bc0 < Nc ? bc0 : Nc;
    int bc1 = nt * 32 + 16 + ml;  bc1 = bc1 < Nc ? bc1 : Nc;
    const bf16* pa0 = A  + (long)ar0 * KP + 8 * g;
    const bf16* pa1 = A  + (long)ar1 * KP + 8 * g;
    const bf16* pb0 = BT + (long)bc0 * KP + 8 * g;
    const bf16* pb1 = BT + (long)bc1 * KP + 8 * g;

    v8f acc00 = {}, acc01 = {}, acc10 = {}, acc11 = {};
    for (int k0 = 0; k0 < KP; k0 += 32) {
        __builtin_prefetch(pa0 + k0 + 64, 0, 1);   // speculative global_prefetch_b8
        __builtin_prefetch(pb0 + k0 + 64, 0, 1);
        v16bf a0 = ldfrag(pa0 + k0);
        v16bf a1 = ldfrag(pa1 + k0);
        v16bf b0 = ldfrag(pb0 + k0);
        v16bf b1 = ldfrag(pb1 + k0);
        acc00 = __builtin_amdgcn_wmma_f32_16x16x32_bf16(false, a0, false, b0, (short)0, acc00, false, false);
        acc01 = __builtin_amdgcn_wmma_f32_16x16x32_bf16(false, a0, false, b1, (short)0, acc01, false, false);
        acc10 = __builtin_amdgcn_wmma_f32_16x16x32_bf16(false, a1, false, b0, (short)0, acc10, false, false);
        acc11 = __builtin_amdgcn_wmma_f32_16x16x32_bf16(false, a1, false, b1, (short)0, acc11, false, false);
    }

    const int cols[2] = { nt * 32 + ml, nt * 32 + 16 + ml };
    const v8f* accs[4] = { &acc00, &acc01, &acc10, &acc11 };
#pragma unroll
    for (int mi = 0; mi < 2; ++mi) {
#pragma unroll
        for (int ni = 0; ni < 2; ++ni) {
            const v8f& acc = *accs[mi * 2 + ni];
            const int col = cols[ni];
#pragma unroll
            for (int r = 0; r < 8; ++r) {
                // C/D layout: VGPR r -> row r (lanes 0-15) / r+8 (lanes 16-31)
                const int row = mt * 32 + mi * 16 + r + 8 * g;
                if (row < M && col < N) {
                    float v = acc[r];
                    if (biasb)   v += biasb[col];
                    if (epi & 2) v = bn1(v, bng[col], bnb[col], bnm[col], bnv[col]);
                    if (epi & 1) v = fmaxf(v, 0.0f);
                    C[(long)row * ldc + col] = v;
                }
            }
        }
    }
}

// ---------------------------------------------------------------- host
extern "C" void kernel_launch(void* const* d_in, const int* in_sizes, int n_in,
                              void* d_out, int out_size, void* d_ws, size_t ws_size,
                              hipStream_t stream) {
    (void)n_in; (void)ws_size;
    const int N = in_sizes[0];
    const int E = in_sizes[2];
    const int G = out_size;                 // head output is (num_graphs, 1)

    const int* xidx  = (const int*)d_in[0];
    const int* ei    = (const int*)d_in[1];
    const int* src   = ei;
    const int* dst   = ei + E;
    const int* eatt  = (const int*)d_in[2];
    const int* batch = (const int*)d_in[3];
    const float* node_emb = (const float*)d_in[5];
    const float* edge_emb = (const float*)d_in[6];
    const float* W_edge = (const float*)d_in[7];
    const float* b_edge = (const float*)d_in[8];
    const float* W_pre  = (const float*)d_in[9];
    const float* b_pre  = (const float*)d_in[10];
    const float* W_post = (const float*)d_in[11];
    const float* b_post = (const float*)d_in[12];
    const float* W_lin  = (const float*)d_in[13];
    const float* b_lin  = (const float*)d_in[14];
    const float* bng = (const float*)d_in[15];
    const float* bnb = (const float*)d_in[16];
    const float* bnm = (const float*)d_in[17];
    const float* bnv = (const float*)d_in[18];
    const float* hw0 = (const float*)d_in[19];
    const float* hb0 = (const float*)d_in[20];
    const float* g0g = (const float*)d_in[21];
    const float* g0b = (const float*)d_in[22];
    const float* g0m = (const float*)d_in[23];
    const float* g0v = (const float*)d_in[24];
    const float* hw1 = (const float*)d_in[25];
    const float* hb1 = (const float*)d_in[26];
    const float* g1g = (const float*)d_in[27];
    const float* g1b = (const float*)d_in[28];
    const float* g1m = (const float*)d_in[29];
    const float* g1v = (const float*)d_in[30];
    const float* hw2 = (const float*)d_in[31];
    const float* hb2 = (const float*)d_in[32];

    // ---- workspace bump allocator
    char* ws = (char*)d_ws;
    size_t off = 0;
    auto alloc = [&](size_t bytes) -> void* {
        void* p = ws + off;
        off = (off + bytes + 255) & ~(size_t)255;
        return p;
    };
    float* h     = (float*)alloc((size_t)N * EMB * 4);
    float* hnew  = (float*)alloc((size_t)N * EMB * 4);
    float* nrep  = (float*)alloc((size_t)N * EMB * 4);
    float* out300= (float*)alloc((size_t)N * EMB * 4);
    bf16*  h_bf  = (bf16*) alloc((size_t)N * KPE * 2);
    bf16*  out_bf= (bf16*) alloc((size_t)N * KPE * 2);
    float* e_f   = (float*)alloc((size_t)E * EMB * 4);
    bf16*  e_bf  = (bf16*) alloc((size_t)E * KPE * 2);
    float* eproj = (float*)alloc((size_t)E * EMB * 4);
    bf16*  hcat  = (bf16*) alloc((size_t)E * KP3 * 2);
    float* m     = (float*)alloc((size_t)TOWERS * E * EMB * 4);
    const size_t TN3 = (size_t)TOWERS * N * EMB;
    float* ssum  = (float*)alloc(TN3 * 4);
    float* smin  = (float*)alloc(TN3 * 4);
    float* smax  = (float*)alloc(TN3 * 4);
    float* ssq   = (float*)alloc(TN3 * 4);
    bf16*  agg   = (bf16*) alloc((size_t)TOWERS * N * AGGP * 2);
    const long TNF = (long)TOWERS * N * FO;
    float* tmp4  = (float*)alloc((size_t)4 * TNF * 4);
    float* cnt   = (float*)alloc((size_t)N * 4);
    float* deg   = (float*)alloc((size_t)N * 4);
    float* ampv  = (float*)alloc((size_t)N * 4);
    float* attv  = (float*)alloc((size_t)N * 4);
    float* hnb   = (float*)alloc((size_t)N * 4);
    float* scal  = (float*)alloc(4 * 4);
    float* gbuf  = (float*)alloc((size_t)G * EMB * 4);
    float* gh1   = (float*)alloc((size_t)G * 2 * EMB * 4);
    float* gh2   = (float*)alloc((size_t)G * EMB * 4);
    bf16*  g_bf  = (bf16*) alloc((size_t)G * KP2E * 2);
    // transposed + K-padded bf16 weights
    bf16*  We_bf = (bf16*) alloc((size_t)NLAYERS * EMB * KPE * 2);
    bf16*  Wp_bf = (bf16*) alloc((size_t)NLAYERS * TOWERS * EMB * KP3 * 2);
    bf16*  Wo0   = (bf16*) alloc((size_t)NLAYERS * TOWERS * FO * KPE * 2);
    bf16*  Wo1   = (bf16*) alloc((size_t)NLAYERS * TOWERS * FO * AGGP * 2);
    bf16*  Wo2   = (bf16*) alloc((size_t)NLAYERS * TOWERS * FO * AGGP * 2);
    bf16*  Wo3   = (bf16*) alloc((size_t)NLAYERS * TOWERS * FO * AGGP * 2);
    bf16*  Wl_bf = (bf16*) alloc((size_t)NLAYERS * EMB * KPE * 2);
    bf16*  h0_bf = (bf16*) alloc((size_t)2 * EMB * KPE * 2);
    bf16*  h1_bf = (bf16*) alloc((size_t)EMB * KP2E * 2);
    bf16*  h2_bf = (bf16*) alloc((size_t)KPE * 2);

    auto blocks = [](long n) -> int {
        long b = (n + 255) / 256;
        return (int)(b < 16384 ? (b > 0 ? b : 1) : 16384);
    };
    auto convpad = [&](const float* in, bf16* o, int K, int KP, long rows) {
        k_f2bf_pad<<<blocks(rows * KP), 256, 0, stream>>>(in, o, K, KP, rows * KP);
    };
    auto wconvT = [&](const float* W, bf16* o, int Ksrc, int rowOff, int K, int Ncol,
                      int KP, long bat) {
        long tot = bat * (long)Ncol * KP;
        k_wconvT<<<blocks(tot), 256, 0, stream>>>(W, o, Ksrc, rowOff, K, Ncol, KP, tot);
    };
    auto gemm = [&](const bf16* A, long sA, const bf16* BT, long sB,
                    const float* bias, long sBias, float* C, long sC, int ldc,
                    int M, int Ncol, int KP, int bat, int epi,
                    const float* gg = nullptr, const float* gb = nullptr,
                    const float* gm = nullptr, const float* gv = nullptr) {
        dim3 blk(32 * GEMM_WAVES);
        dim3 grd((Ncol + 31) / 32, (M + 32 * GEMM_WAVES - 1) / (32 * GEMM_WAVES), bat);
        k_gemm_bf16<<<grd, blk, 0, stream>>>(A, sA, BT, sB, bias, sBias,
                                             C, sC, ldc, M, Ncol, KP, epi, gg, gb, gm, gv);
    };

    // ---- weight conversion (transpose + pad, bf16)
    wconvT(W_edge, We_bf, EMB, 0, EMB, EMB, KPE, NLAYERS);
    wconvT(W_pre,  Wp_bf, 3 * EMB, 0, 3 * EMB, EMB, KP3, (long)NLAYERS * TOWERS);
    wconvT(W_post, Wo0, 13 * EMB, 0,        EMB,  FO, KPE,  (long)NLAYERS * TOWERS);
    wconvT(W_post, Wo1, 13 * EMB, EMB,      AGGW, FO, AGGP, (long)NLAYERS * TOWERS);
    wconvT(W_post, Wo2, 13 * EMB, 5 * EMB,  AGGW, FO, AGGP, (long)NLAYERS * TOWERS);
    wconvT(W_post, Wo3, 13 * EMB, 9 * EMB,  AGGW, FO, AGGP, (long)NLAYERS * TOWERS);
    wconvT(W_lin,  Wl_bf, EMB, 0, EMB, EMB, KPE, NLAYERS);
    wconvT(hw0, h0_bf, EMB, 0, EMB, 2 * EMB, KPE, 1);
    wconvT(hw1, h1_bf, 2 * EMB, 0, 2 * EMB, EMB, KP2E, 1);
    wconvT(hw2, h2_bf, EMB, 0, EMB, 1, KPE, 1);

    // ---- embeddings + degree stats
    k_embed<<<blocks((long)N * EMB), 256, 0, stream>>>(xidx, node_emb, h, (long)N * EMB);
    k_embed<<<blocks((long)E * EMB), 256, 0, stream>>>(eatt, edge_emb, e_f, (long)E * EMB);
    convpad(e_f, e_bf, EMB, KPE, E);
    hipMemcpyAsync(nrep, h, (size_t)N * EMB * 4, hipMemcpyDeviceToDevice, stream);
    k_fill<<<blocks(N), 256, 0, stream>>>(cnt, N, 0.0f);
    k_fill<<<1, 32, 0, stream>>>(scal, 4, 0.0f);
    k_count<<<blocks(E), 256, 0, stream>>>(dst, cnt, E);
    k_logsum<<<blocks(N), 256, 0, stream>>>(cnt, scal, N);
    k_node_stats<<<blocks(N), 256, 0, stream>>>(cnt, scal, deg, ampv, attv, hnb, N);

    // ---- layers
    for (int l = 0; l < NLAYERS; ++l) {
        convpad(h, h_bf, EMB, KPE, N);

        // eproj = e @ W_edge[l] + b_edge[l]
        gemm(e_bf, 0, We_bf + (long)l * EMB * KPE, 0,
             b_edge + (long)l * EMB, 0, eproj, 0, EMB, E, EMB, KPE, 1, 0);

        k_build_hcat<<<blocks((long)E * KP3), 256, 0, stream>>>(h, src, dst, eproj, hcat,
                                                                (long)E * KP3);

        // m[t] = hcat @ W_pre[l,t] + b_pre[l,t]   (batched over towers)
        gemm(hcat, 0, Wp_bf + (long)l * TOWERS * EMB * KP3, (long)EMB * KP3,
             b_pre + (long)l * TOWERS * EMB, EMB,
             m, (long)E * EMB, EMB, E, EMB, KP3, TOWERS, 0);

        // segment reductions over dst
        k_fill<<<blocks((long)TN3), 256, 0, stream>>>(ssum, (long)TN3, 0.0f);
        k_fill<<<blocks((long)TN3), 256, 0, stream>>>(ssq,  (long)TN3, 0.0f);
        k_fill<<<blocks((long)TN3), 256, 0, stream>>>(smin, (long)TN3, 3.0e38f);
        k_fill<<<blocks((long)TN3), 256, 0, stream>>>(smax, (long)TN3, -3.0e38f);
        k_scatter_stats<<<blocks((long)TOWERS * E * EMB), 256, 0, stream>>>(
            m, dst, ssum, smin, smax, ssq, E, N, (long)TOWERS * E * EMB);
        k_finalize_agg<<<blocks((long)TOWERS * N * AGGP), 256, 0, stream>>>(
            ssum, smin, smax, ssq, deg, hnb, agg, N, (long)TOWERS * N * AGGP);

        // post einsum split over W_post rows: [x | agg | agg*amp | agg*att]
        gemm(h_bf, 0, Wo0 + (long)l * TOWERS * FO * KPE, (long)FO * KPE, nullptr, 0,
             tmp4, (long)N * FO, FO, N, FO, KPE, TOWERS, 0);
        gemm(agg, (long)N * AGGP, Wo1 + (long)l * TOWERS * FO * AGGP, (long)FO * AGGP,
             nullptr, 0, tmp4 + TNF, (long)N * FO, FO, N, FO, AGGP, TOWERS, 0);
        gemm(agg, (long)N * AGGP, Wo2 + (long)l * TOWERS * FO * AGGP, (long)FO * AGGP,
             nullptr, 0, tmp4 + 2 * TNF, (long)N * FO, FO, N, FO, AGGP, TOWERS, 0);
        gemm(agg, (long)N * AGGP, Wo3 + (long)l * TOWERS * FO * AGGP, (long)FO * AGGP,
             nullptr, 0, tmp4 + 3 * TNF, (long)N * FO, FO, N, FO, AGGP, TOWERS, 0);
        k_combine_post<<<blocks((long)N * TOWERS * FO), 256, 0, stream>>>(
            tmp4, ampv, attv, b_post + (long)l * TOWERS * FO, out300, N,
            (long)N * TOWERS * FO);

        // hnew = BN(out300 @ W_lin[l] + b_lin[l]) [+relu if not last layer]
        convpad(out300, out_bf, EMB, KPE, N);
        int epi = (l < NLAYERS - 1) ? 3 : 2;
        gemm(out_bf, 0, Wl_bf + (long)l * EMB * KPE, 0,
             b_lin + (long)l * EMB, 0, hnew, 0, EMB, N, EMB, KPE, 1, epi,
             bng + (long)l * EMB, bnb + (long)l * EMB,
             bnm + (long)l * EMB, bnv + (long)l * EMB);

        if (l < NLAYERS - 1)   // node_rep sums h_list[0..LAYERS-1] only
            k_add<<<blocks((long)N * EMB), 256, 0, stream>>>(nrep, hnew, (long)N * EMB);
        float* t = h; h = hnew; hnew = t;
    }

    // ---- readout + head
    k_fill<<<blocks((long)G * EMB), 256, 0, stream>>>(gbuf, (long)G * EMB, 0.0f);
    k_segsum<<<blocks((long)N * EMB), 256, 0, stream>>>(nrep, batch, gbuf, (long)N * EMB);

    convpad(gbuf, g_bf, EMB, KPE, G);
    gemm(g_bf, 0, h0_bf, 0, hb0, 0, gh1, 0, 2 * EMB,
         G, 2 * EMB, KPE, 1, 3, g0g, g0b, g0m, g0v);
    convpad(gh1, g_bf, 2 * EMB, KP2E, G);
    gemm(g_bf, 0, h1_bf, 0, hb1, 0, gh2, 0, EMB,
         G, EMB, KP2E, 1, 3, g1g, g1b, g1m, g1v);
    convpad(gh2, g_bf, EMB, KPE, G);
    gemm(g_bf, 0, h2_bf, 0, hb2, 0, (float*)d_out, 0, 1,
         G, 1, KPE, 1, 0);
}
